// FusedMoE_68152541053515
// MI455X (gfx1250) — matrix operand order repaired and checked by reading, not compiled, not run on hardware
//
#include <hip/hip_runtime.h>
#include <hip/hip_bf16.h>

#define TT 4096
#define HH 2048
#define II 1408
#define EE 8
#define KK 2
#define CAP (TT*KK)        // worst-case pairs per expert
#define BM  128            // rows per block (8 waves, one 16-row wmma tile each)
#define MT  (CAP/BM)       // 64 m-blocks per expert (worst case)
#define NB1 (II/64)        // 22 n-blocks for gemm1 (64 gate + 64 up cols each)
#define NB2 (HH/64)        // 32 n-blocks for gemm2 (64 cols each)
#define LDA 48             // padded LDS stride in halves for a 32-k tile (96B)

typedef __attribute__((ext_vector_type(16))) __bf16 v16bf;
typedef __attribute__((ext_vector_type(8)))  float  v8f;

union Frag { uint4 u[2]; v16bf v; };

// scheduling hints: interleave DS-reads (0x100) with WMMA (0x8)
#if __has_builtin(__builtin_amdgcn_sched_group_barrier)
#define SGB(mask, n) __builtin_amdgcn_sched_group_barrier(mask, n, 0)
#else
#define SGB(mask, n)
#endif

// fp32 pair -> packed bf16 (low = a, high = b); single VALU op on gfx1250
static __device__ __forceinline__ unsigned pack2(float a, float b){
  unsigned r;
  asm("v_cvt_pk_bf16_f32 %0, %1, %2" : "=v"(r) : "v"(a), "v"(b));
  return r;
}
static __device__ __forceinline__ __bf16 f2bf(float f){
  union { unsigned u; __bf16 b[2]; } y; y.u = pack2(f, f); return y.b[0];
}

static __device__ __forceinline__ v8f wmma_bf16(const Frag& a, const Frag& b, v8f c){
  return __builtin_amdgcn_wmma_f32_16x16x32_bf16(false, a.v, false, b.v, (short)0, c, false, false);
}

// ---- CDNA5 async global->LDS (ASYNCcnt) ----
static __device__ __forceinline__ void async_b128(unsigned lds, unsigned voff, const void* sbase){
  asm volatile("global_load_async_to_lds_b128 %0, %1, %2"
               :: "v"(lds), "v"(voff), "s"(sbase) : "memory");
}
template<int N> static __device__ __forceinline__ void wait_async(){
  asm volatile("s_wait_asynccnt %0" :: "i"(N) : "memory");
}
static __device__ __forceinline__ unsigned lds_off(const void* p){
  return (unsigned)(uintptr_t)p;   // low 32 bits of generic LDS address = LDS byte offset
}

// ---------------- generic fp32 -> bf16 cast (x, w13, w2) ----------------
__global__ __launch_bounds__(256) void cast_k(const float* __restrict__ x,
                                              __bf16* __restrict__ y){
  size_t i = ((size_t)blockIdx.x * 256 + threadIdx.x) * 4;
  float4 v = *(const float4*)(x + i);
  uint2 p; p.x = pack2(v.x, v.y); p.y = pack2(v.z, v.w);
  *(uint2*)(y + i) = p;
}

// ---------------- build per-expert token buckets ----------------
__global__ __launch_bounds__(256) void route(const int* __restrict__ ids,
                                             int* __restrict__ cnt,
                                             int* __restrict__ bucket){
  int p = blockIdx.x * 256 + threadIdx.x;   // pair id in [0, T*K)
  int e = ids[p];
  int pos = atomicAdd(&cnt[e], 1);
  bucket[e * CAP + pos] = p;
}

// ================= fast path: bf16 weights, fully async double-buffered A+B =================
__global__ __launch_bounds__(256) void gemm1_async(
    const __bf16* __restrict__ w13b, const float* __restrict__ tw,
    const __bf16* __restrict__ xb, const int* __restrict__ cnt,
    const int* __restrict__ bucket, __bf16* __restrict__ act)
{
  int blk = blockIdx.x;
  int e  = blk / (MT * NB1);
  int r  = blk % (MT * NB1);
  int mt = r / NB1;
  int nb = r % NB1;
  int c  = cnt[e];
  int m0 = mt * BM;
  if (m0 >= c) return;

  __shared__ __align__(16) __bf16 As[2][BM  * LDA];  // 128 token rows x 32 k
  __shared__ __align__(16) __bf16 Bs[2][128 * LDA];  // 64 gate + 64 up cols x 32 k
  __shared__ int   prs[BM];
  __shared__ float rws[BM];

  int tid = threadIdx.x;
  if (tid < BM){
    int row = m0 + tid;
    int p = bucket[e * CAP + (row < c ? row : m0)];
    prs[tid] = p;
    rws[tid] = (row < c) ? tw[p] : 0.f;
  }
  __syncthreads();   // prs/rws visible

  int wave = tid >> 5, lane = tid & 31, lr = lane & 15, hi = lane >> 4;

  // per-thread DMA assignments: one 16B chunk each of 2 A rows + 1 gate col + 1 up col
  int r0 = tid >> 2, seg = tid & 3;                  // r0 in 0..63; rows r0, r0+64
  unsigned voffA0 = ((unsigned)(prs[r0]      >> 1) * (unsigned)HH + seg * 8) * 2;
  unsigned voffA1 = ((unsigned)(prs[r0 + 64] >> 1) * (unsigned)HH + seg * 8) * 2;
  const __bf16* wb = w13b + (size_t)e * (2 * II) * HH;
  unsigned voffG = ((unsigned)(nb * 64 + r0)      * (unsigned)HH + seg * 8) * 2;
  unsigned voffU = ((unsigned)(II + nb * 64 + r0) * (unsigned)HH + seg * 8) * 2;
  unsigned ldsA0[2], ldsA1[2], ldsG[2], ldsU[2];
  #pragma unroll
  for (int q = 0; q < 2; ++q){
    ldsA0[q] = lds_off(&As[q][r0 * LDA + seg * 8]);
    ldsA1[q] = lds_off(&As[q][(r0 + 64) * LDA + seg * 8]);
    ldsG[q]  = lds_off(&Bs[q][r0 * LDA + seg * 8]);
    ldsU[q]  = lds_off(&Bs[q][(64 + r0) * LDA + seg * 8]);
  }

  v8f ag[4] = {{}, {}, {}, {}};
  v8f au[4] = {{}, {}, {}, {}};

  const int STEPS = HH / 32;   // 64
  // prologue: DMA first tiles
  async_b128(ldsA0[0], voffA0, xb);
  async_b128(ldsA1[0], voffA1, xb);
  async_b128(ldsG[0],  voffG,  wb);
  async_b128(ldsU[0],  voffU,  wb);

  for (int i = 0; i < STEPS; ++i){
    int b = i & 1;
    if (i + 1 < STEPS){                    // DMA next tiles (buffers protected by trailing barrier)
      unsigned off = (unsigned)((i + 1) * 32 * 2);
      async_b128(ldsA0[b ^ 1], voffA0 + off, xb);
      async_b128(ldsA1[b ^ 1], voffA1 + off, xb);
      async_b128(ldsG[b ^ 1],  voffG  + off, wb);
      async_b128(ldsU[b ^ 1],  voffU  + off, wb);
      wait_async<4>();                     // batch i complete (4 outstanding = batch i+1)
    } else {
      wait_async<0>();
    }
    __syncthreads();

    const __bf16* A = As[b];
    const __bf16* B = Bs[b];
    Frag a;
    a.u[0] = *(const uint4*)&A[(16 * wave + lr) * LDA + hi * 16];
    a.u[1] = *(const uint4*)&A[(16 * wave + lr) * LDA + hi * 16 + 8];
    #pragma unroll
    for (int t = 0; t < 4; ++t){
      Frag bg, bu;
      bg.u[0] = *(const uint4*)&B[(16 * t + lr) * LDA + hi * 16];
      bg.u[1] = *(const uint4*)&B[(16 * t + lr) * LDA + hi * 16 + 8];
      bu.u[0] = *(const uint4*)&B[(64 + 16 * t + lr) * LDA + hi * 16];
      bu.u[1] = *(const uint4*)&B[(64 + 16 * t + lr) * LDA + hi * 16 + 8];
      ag[t] = wmma_bf16(a, bg, ag[t]);
      au[t] = wmma_bf16(a, bu, au[t]);
    }
    // pin schedule: 18 DS reads vs 8 WMMA -> keep 2-3 fragment buffers in flight
    SGB(0x100, 6);                          // A pair + 2 B pairs
    #pragma unroll
    for (int t = 0; t < 6; ++t){ SGB(0x8, 1); SGB(0x100, 2); }
    SGB(0x8, 2);
    __syncthreads();
  }

  #pragma unroll
  for (int t = 0; t < 4; ++t){
    int icol = nb * 64 + 16 * t + lr;
    #pragma unroll
    for (int v = 0; v < 8; ++v){
      int row = 16 * wave + 8 * hi + v;
      if (m0 + row < c){
        float gv = ag[t][v], uv = au[t][v];
        float s = gv * __builtin_amdgcn_rcpf(1.f + __expf(-gv));
        act[(size_t)prs[row] * II + icol] = f2bf(s * uv * rws[row]);
      }
    }
  }
}

__global__ __launch_bounds__(256) void gemm2_async(
    const __bf16* __restrict__ w2b, const __bf16* __restrict__ act,
    const int* __restrict__ cnt, const int* __restrict__ bucket,
    float* __restrict__ out)
{
  int blk = blockIdx.x;
  int e  = blk / (MT * NB2);
  int r  = blk % (MT * NB2);
  int mt = r / NB2;
  int nb = r % NB2;
  int c  = cnt[e];
  int m0 = mt * BM;
  if (m0 >= c) return;

  __shared__ __align__(16) __bf16 As[2][BM * LDA];
  __shared__ __align__(16) __bf16 Bs[2][64 * LDA];
  __shared__ int prs[BM];

  int tid = threadIdx.x;
  if (tid < BM){
    int row = m0 + tid;
    prs[tid] = bucket[e * CAP + (row < c ? row : m0)];
  }
  __syncthreads();

  int wave = tid >> 5, lane = tid & 31, lr = lane & 15, hi = lane >> 4;

  int r0 = tid >> 2, seg = tid & 3;
  unsigned voffA0 = ((unsigned)prs[r0]      * (unsigned)II + seg * 8) * 2;
  unsigned voffA1 = ((unsigned)prs[r0 + 64] * (unsigned)II + seg * 8) * 2;
  const __bf16* wb = w2b + (size_t)e * HH * II;
  unsigned voffB = ((unsigned)(nb * 64 + r0) * (unsigned)II + seg * 8) * 2;
  unsigned ldsA0[2], ldsA1[2], ldsB[2];
  #pragma unroll
  for (int q = 0; q < 2; ++q){
    ldsA0[q] = lds_off(&As[q][r0 * LDA + seg * 8]);
    ldsA1[q] = lds_off(&As[q][(r0 + 64) * LDA + seg * 8]);
    ldsB[q]  = lds_off(&Bs[q][r0 * LDA + seg * 8]);
  }

  v8f acc[4] = {{}, {}, {}, {}};

  const int STEPS = II / 32;   // 44
  async_b128(ldsA0[0], voffA0, act);
  async_b128(ldsA1[0], voffA1, act);
  async_b128(ldsB[0],  voffB,  wb);

  for (int i = 0; i < STEPS; ++i){
    int b = i & 1;
    if (i + 1 < STEPS){
      unsigned off = (unsigned)((i + 1) * 32 * 2);
      async_b128(ldsA0[b ^ 1], voffA0 + off, act);
      async_b128(ldsA1[b ^ 1], voffA1 + off, act);
      async_b128(ldsB[b ^ 1],  voffB  + off, wb);
      wait_async<3>();
    } else {
      wait_async<0>();
    }
    __syncthreads();

    const __bf16* A = As[b];
    const __bf16* B = Bs[b];
    Frag a;
    a.u[0] = *(const uint4*)&A[(16 * wave + lr) * LDA + hi * 16];
    a.u[1] = *(const uint4*)&A[(16 * wave + lr) * LDA + hi * 16 + 8];
    #pragma unroll
    for (int t = 0; t < 4; ++t){
      Frag bb;
      bb.u[0] = *(const uint4*)&B[(16 * t + lr) * LDA + hi * 16];
      bb.u[1] = *(const uint4*)&B[(16 * t + lr) * LDA + hi * 16 + 8];
      acc[t] = wmma_bf16(a, bb, acc[t]);
    }
    // pin schedule: 10 DS reads vs 4 WMMA
    SGB(0x100, 6);
    #pragma unroll
    for (int t = 0; t < 2; ++t){ SGB(0x8, 1); SGB(0x100, 2); }
    SGB(0x8, 2);
    __syncthreads();
  }

  #pragma unroll
  for (int t = 0; t < 4; ++t){
    int hcol = nb * 64 + 16 * t + lr;
    #pragma unroll
    for (int v = 0; v < 8; ++v){
      int row = 16 * wave + 8 * hi + v;
      if (m0 + row < c){
        int tok = prs[row] >> 1;
        atomicAdd(&out[(size_t)tok * HH + hcol], acc[t][v]);
      }
    }
  }
}

// ================= fallback path: fp32 weights converted in-loop =================
__global__ __launch_bounds__(256) void gemm1_cvt(
    const float* __restrict__ w13, const float* __restrict__ tw,
    const __bf16* __restrict__ xb, const int* __restrict__ cnt,
    const int* __restrict__ bucket, __bf16* __restrict__ act)
{
  int blk = blockIdx.x;
  int e  = blk / (MT * NB1);
  int r  = blk % (MT * NB1);
  int mt = r / NB1;
  int nb = r % NB1;
  int c  = cnt[e];
  int m0 = mt * BM;
  if (m0 >= c) return;

  __shared__ __align__(16) __bf16 Bs[128 * LDA];
  __shared__ int   prs[BM];
  __shared__ float rws[BM];

  int tid = threadIdx.x;
  if (tid < BM){
    int row = m0 + tid;
    int p = bucket[e * CAP + (row < c ? row : m0)];
    prs[tid] = p;
    rws[tid] = (row < c) ? tw[p] : 0.f;
  }
  __syncthreads();

  int wave = tid >> 5, lane = tid & 31, lr = lane & 15, hi = lane >> 4;
  const __bf16* abase = xb + (size_t)(prs[16 * wave + lr] >> 1) * HH + hi * 16;
  int bc = tid >> 3, bchunk = tid & 7;
  const float* wb = w13 + (size_t)e * (2 * II) * HH;

  v8f ag[4] = {{}, {}, {}, {}};
  v8f au[4] = {{}, {}, {}, {}};

  for (int h0 = 0; h0 < HH; h0 += 32){
    #pragma unroll
    for (int j = 0; j < 4; ++j){
      int col = bc + 32 * j;
      int grow = (col < 64) ? (nb * 64 + col) : (II + nb * 64 + (col - 64));
      float4 wv = *(const float4*)(wb + (size_t)grow * HH + h0 + bchunk * 4);
      uint2 pk; pk.x = pack2(wv.x, wv.y); pk.y = pack2(wv.z, wv.w);
      *(uint2*)&Bs[col * LDA + bchunk * 4] = pk;
    }
    Frag a;
    a.u[0] = *(const uint4*)(abase + h0);
    a.u[1] = *(const uint4*)(abase + h0 + 8);
    __syncthreads();

    #pragma unroll
    for (int t = 0; t < 4; ++t){
      Frag bg, bu;
      bg.u[0] = *(const uint4*)&Bs[(16 * t + lr) * LDA + hi * 16];
      bg.u[1] = *(const uint4*)&Bs[(16 * t + lr) * LDA + hi * 16 + 8];
      bu.u[0] = *(const uint4*)&Bs[(64 + 16 * t + lr) * LDA + hi * 16];
      bu.u[1] = *(const uint4*)&Bs[(64 + 16 * t + lr) * LDA + hi * 16 + 8];
      ag[t] = wmma_bf16(a, bg, ag[t]);
      au[t] = wmma_bf16(a, bu, au[t]);
    }
    __syncthreads();
  }

  #pragma unroll
  for (int t = 0; t < 4; ++t){
    int icol = nb * 64 + 16 * t + lr;
    #pragma unroll
    for (int v = 0; v < 8; ++v){
      int row = 16 * wave + 8 * hi + v;
      if (m0 + row < c){
        float gv = ag[t][v], uv = au[t][v];
        float s = gv * __builtin_amdgcn_rcpf(1.f + __expf(-gv));
        act[(size_t)prs[row] * II + icol] = f2bf(s * uv * rws[row]);
      }
    }
  }
}

__global__ __launch_bounds__(256) void gemm2_cvt(
    const float* __restrict__ w2, const __bf16* __restrict__ act,
    const int* __restrict__ cnt, const int* __restrict__ bucket,
    float* __restrict__ out)
{
  int blk = blockIdx.x;
  int e  = blk / (MT * NB2);
  int r  = blk % (MT * NB2);
  int mt = r / NB2;
  int nb = r % NB2;
  int c  = cnt[e];
  int m0 = mt * BM;
  if (m0 >= c) return;

  __shared__ __align__(16) __bf16 Bs[64 * LDA];
  __shared__ int prs[BM];

  int tid = threadIdx.x;
  if (tid < BM){
    int row = m0 + tid;
    prs[tid] = bucket[e * CAP + (row < c ? row : m0)];
  }
  __syncthreads();

  int wave = tid >> 5, lane = tid & 31, lr = lane & 15, hi = lane >> 4;
  const __bf16* abase = act + (size_t)prs[16 * wave + lr] * II + hi * 16;
  int bc = tid >> 3, bchunk = tid & 7;
  const float* wb = w2 + (size_t)e * HH * II;

  v8f acc[4] = {{}, {}, {}, {}};

  for (int k0 = 0; k0 < II; k0 += 32){
    #pragma unroll
    for (int j = 0; j < 2; ++j){
      int col = bc + 32 * j;
      int h = nb * 64 + col;
      float4 wv = *(const float4*)(wb + (size_t)h * II + k0 + bchunk * 4);
      uint2 pk; pk.x = pack2(wv.x, wv.y); pk.y = pack2(wv.z, wv.w);
      *(uint2*)&Bs[col * LDA + bchunk * 4] = pk;
    }
    Frag a;
    a.u[0] = *(const uint4*)(abase + k0);
    a.u[1] = *(const uint4*)(abase + k0 + 8);
    __syncthreads();

    #pragma unroll
    for (int t = 0; t < 4; ++t){
      Frag b;
      b.u[0] = *(const uint4*)&Bs[(16 * t + lr) * LDA + hi * 16];
      b.u[1] = *(const uint4*)&Bs[(16 * t + lr) * LDA + hi * 16 + 8];
      acc[t] = wmma_bf16(a, b, acc[t]);
    }
    __syncthreads();
  }

  #pragma unroll
  for (int t = 0; t < 4; ++t){
    int hcol = nb * 64 + 16 * t + lr;
    #pragma unroll
    for (int v = 0; v < 8; ++v){
      int row = 16 * wave + 8 * hi + v;
      if (m0 + row < c){
        int tok = prs[row] >> 1;
        atomicAdd(&out[(size_t)tok * HH + hcol], acc[t][v]);
      }
    }
  }
}

extern "C" void kernel_launch(void* const* d_in, const int* in_sizes, int n_in,
                              void* d_out, int out_size, void* d_ws, size_t ws_size,
                              hipStream_t stream) {
  const float* hs  = (const float*)d_in[0];   // (T,H) fp32
  const float* w13 = (const float*)d_in[1];   // (E,2I,H) fp32
  const float* w2  = (const float*)d_in[2];   // (E,H,I) fp32
  const int*   ids = (const int*)d_in[3];     // (T,K) int32
  const float* tw  = (const float*)d_in[4];   // (T,K) fp32
  float* out = (float*)d_out;                 // (T,H) fp32

  const size_t n_w13 = (size_t)EE * 2 * II * HH;   // 46,137,344
  const size_t n_w2  = (size_t)EE * HH * II;       // 23,068,672

  char* ws = (char*)d_ws;
  size_t o = 0;
  __bf16* xb  = (__bf16*)(ws + o); o += (size_t)TT * HH * 2;        // 16 MB
  __bf16* act = (__bf16*)(ws + o); o += (size_t)TT * KK * II * 2;   // 22 MB
  o = (o + 255) & ~(size_t)255;
  int* cnt    = (int*)(ws + o);    o += 256;
  int* bucket = (int*)(ws + o);    o += (size_t)EE * CAP * 4;       // 256 KB
  o = (o + 255) & ~(size_t)255;
  __bf16* w13b = (__bf16*)(ws + o); size_t o_full = o + n_w13 * 2;  // +92 MB
  __bf16* w2b  = (__bf16*)(ws + o_full); o_full += n_w2 * 2;        // +46 MB
  const bool full = (ws_size >= o_full);

  hipMemsetAsync(cnt, 0, EE * sizeof(int), stream);
  hipMemsetAsync(out, 0, (size_t)TT * HH * sizeof(float), stream);

  cast_k<<<(TT * HH) / 1024, 256, 0, stream>>>(hs, xb);
  route <<<(TT * KK) / 256,  256, 0, stream>>>(ids, cnt, bucket);

  if (full){
    cast_k<<<(int)(n_w13 / 1024), 256, 0, stream>>>(w13, w13b);
    cast_k<<<(int)(n_w2  / 1024), 256, 0, stream>>>(w2,  w2b);
    gemm1_async<<<EE * MT * NB1, 256, 0, stream>>>(w13b, tw, xb, cnt, bucket, act);
    gemm2_async<<<EE * MT * NB2, 256, 0, stream>>>(w2b, act, cnt, bucket, out);
  } else {
    gemm1_cvt<<<EE * MT * NB1, 256, 0, stream>>>(w13, tw, xb, cnt, bucket, act);
    gemm2_cvt<<<EE * MT * NB2, 256, 0, stream>>>(w2, act, cnt, bucket, out);
  }
}
